// MultiHeadedAttention_11510512353854
// MI455X (gfx1250) — matrix operand order repaired and checked by reading, not compiled, not run on hardware
//
#include <hip/hip_runtime.h>

// ---------------- problem constants (match reference) ----------------
constexpr int B = 2, S = 2048, D = 1024, H = 16, DK = 64;

// ---------------- CDNA5 WMMA types ----------------
typedef __bf16 v16bf __attribute__((ext_vector_type(16)));
typedef float  v8f   __attribute__((ext_vector_type(8)));
typedef unsigned int u32x4 __attribute__((ext_vector_type(4)));

union Frag32B {                 // one WMMA A/B fragment: 16 bf16 = 8 VGPRs
  u32x4 q[2];
  v16bf v;
  unsigned short h[16];
};
union Chunk16B {
  u32x4 q;
  unsigned short h[8];
};

__device__ __forceinline__ unsigned short f2bf(float f) {
  unsigned int u = __float_as_uint(f);
  u += 0x7FFFu + ((u >> 16) & 1u);          // round to nearest even
  return (unsigned short)(u >> 16);
}

__device__ __forceinline__ v8f wmma_bf16(v16bf a, v16bf b, v8f c) {
  // emits v_wmma_f32_16x16x32_bf16
  return __builtin_amdgcn_wmma_f32_16x16x32_bf16(
      /*neg_a=*/false, a, /*neg_b=*/false, b,
      /*c_mod=*/(short)0, c, /*reuse_a=*/false, /*reuse_b=*/false);
}

__device__ __forceinline__ float rmax16(float x) {
#pragma unroll
  for (int m = 1; m < 16; m <<= 1) x = fmaxf(x, __shfl_xor(x, m, 32));
  return x;
}
__device__ __forceinline__ float rsum16(float x) {
#pragma unroll
  for (int m = 1; m < 16; m <<= 1) x += __shfl_xor(x, m, 32);
  return x;
}

// =====================================================================
// GEMM 1: out[b,h,s,dk](bf16) = X(f32, [B*S,D]) @ W(f32,[D,D]) + bias
// block: 128 threads (4 waves), tile M=64,N=64, K-step 32
// =====================================================================
constexpr int AST = 40;   // padded LDS row stride (elements) for 32-wide K tiles

__global__ __launch_bounds__(128)
void gemm_qkv(const float* __restrict__ X, const float* __restrict__ W,
              const float* __restrict__ bias, unsigned short* __restrict__ out) {
  __shared__ __attribute__((aligned(16))) unsigned short As[64 * AST];
  __shared__ __attribute__((aligned(16))) unsigned short Ws[64 * AST];

  const int tid = threadIdx.x;
  const int w   = tid >> 5;
  const int l   = tid & 31;
  const int lm  = l & 15;
  const int kh8 = (l >> 4) * 8;    // A-frag chunk base (ISA 16-bit A layout)
  const int kb16= (l >> 4) * 16;   // unused; B frags also use kh8
  const int m0  = blockIdx.y * 64;
  const int n0  = blockIdx.x * 64;

  v8f acc[4];
  const v8f zf = {0.f,0.f,0.f,0.f,0.f,0.f,0.f,0.f};
#pragma unroll
  for (int nt = 0; nt < 4; ++nt) acc[nt] = zf;

  for (int k0 = 0; k0 < D; k0 += 32) {
    __syncthreads();
    // A tile: 64x32 f32 -> bf16 (row-major)
#pragma unroll
    for (int i = 0; i < 4; ++i) {
      int c  = tid + i * 128;              // 0..511 float4 chunks
      int m  = c >> 3;
      int kk = (c & 7) * 4;
      float4 f = *(const float4*)(X + (size_t)(m0 + m) * D + k0 + kk);
      unsigned long long pk =
          (unsigned long long)f2bf(f.x) |
          ((unsigned long long)f2bf(f.y) << 16) |
          ((unsigned long long)f2bf(f.z) << 32) |
          ((unsigned long long)f2bf(f.w) << 48);
      *(unsigned long long*)&As[m * AST + kk] = pk;
    }
    // W tile: load [k][n] coalesced, store transposed Wt[n][k]
#pragma unroll
    for (int i = 0; i < 4; ++i) {
      int c  = tid + i * 128;
      int kk = c >> 4;                     // 0..31
      int n4 = (c & 15) * 4;
      float4 f = *(const float4*)(W + (size_t)(k0 + kk) * D + n0 + n4);
      Ws[(n4 + 0) * AST + kk] = f2bf(f.x);
      Ws[(n4 + 1) * AST + kk] = f2bf(f.y);
      Ws[(n4 + 2) * AST + kk] = f2bf(f.z);
      Ws[(n4 + 3) * AST + kk] = f2bf(f.w);
    }
    __syncthreads();

    Frag32B a;
    const int mrow = w * 16 + lm;
    a.q[0] = *(const u32x4*)&As[mrow * AST + kh8];
    a.q[1] = *(const u32x4*)&As[mrow * AST + 16 + kh8];
#pragma unroll
    for (int nt = 0; nt < 4; ++nt) {
      Frag32B bfr;
      const int ncol = nt * 16 + lm;
      bfr.q[0] = *(const u32x4*)&Ws[ncol * AST + kh8];
      bfr.q[1] = *(const u32x4*)&Ws[ncol * AST + 16 + kh8];
      acc[nt] = wmma_bf16(a.v, bfr.v, acc[nt]);
    }
  }

  // epilogue: +bias, bf16, scatter to [B,H,S,DK]
#pragma unroll
  for (int nt = 0; nt < 4; ++nt) {
    const int ncol = n0 + nt * 16 + lm;
    const float bv = bias[ncol];
    const int h = ncol / DK, dk = ncol % DK;
#pragma unroll
    for (int i = 0; i < 8; ++i) {
      int r = m0 + w * 16 + i + ((l >> 4) << 3);   // C layout: rows i / i+8
      int bb = r / S, s = r % S;
      out[(((size_t)bb * H + h) * S + s) * DK + dk] = f2bf(acc[nt][i] + bv);
    }
  }
}

// =====================================================================
// GEMM 2: d_out(f32,[B*S,D]) = Xb(bf16,[B*S,D]) @ Wo + bo
// =====================================================================
__global__ __launch_bounds__(128)
void gemm_out(const unsigned short* __restrict__ Xb, const float* __restrict__ W,
              const float* __restrict__ bias, float* __restrict__ out) {
  __shared__ __attribute__((aligned(16))) unsigned short As[64 * AST];
  __shared__ __attribute__((aligned(16))) unsigned short Ws[64 * AST];

  const int tid = threadIdx.x;
  const int w   = tid >> 5;
  const int l   = tid & 31;
  const int lm  = l & 15;
  const int kh8 = (l >> 4) * 8;
  const int kb16= (l >> 4) * 16;
  const int m0  = blockIdx.y * 64;
  const int n0  = blockIdx.x * 64;

  v8f acc[4];
  const v8f zf = {0.f,0.f,0.f,0.f,0.f,0.f,0.f,0.f};
#pragma unroll
  for (int nt = 0; nt < 4; ++nt) acc[nt] = zf;

  for (int k0 = 0; k0 < D; k0 += 32) {
    __syncthreads();
    // A tile already bf16: straight 16B copies
#pragma unroll
    for (int i = 0; i < 2; ++i) {
      int c  = tid + i * 128;              // 0..255 chunks of 8 bf16
      int m  = c >> 2;
      int kk = (c & 3) * 8;
      u32x4 v = *(const u32x4*)(Xb + (size_t)(m0 + m) * D + k0 + kk);
      *(u32x4*)&As[m * AST + kk] = v;
    }
#pragma unroll
    for (int i = 0; i < 4; ++i) {
      int c  = tid + i * 128;
      int kk = c >> 4;
      int n4 = (c & 15) * 4;
      float4 f = *(const float4*)(W + (size_t)(k0 + kk) * D + n0 + n4);
      Ws[(n4 + 0) * AST + kk] = f2bf(f.x);
      Ws[(n4 + 1) * AST + kk] = f2bf(f.y);
      Ws[(n4 + 2) * AST + kk] = f2bf(f.z);
      Ws[(n4 + 3) * AST + kk] = f2bf(f.w);
    }
    __syncthreads();

    Frag32B a;
    const int mrow = w * 16 + lm;
    a.q[0] = *(const u32x4*)&As[mrow * AST + kh8];
    a.q[1] = *(const u32x4*)&As[mrow * AST + 16 + kh8];
#pragma unroll
    for (int nt = 0; nt < 4; ++nt) {
      Frag32B bfr;
      const int ncol = nt * 16 + lm;
      bfr.q[0] = *(const u32x4*)&Ws[ncol * AST + kh8];
      bfr.q[1] = *(const u32x4*)&Ws[ncol * AST + 16 + kh8];
      acc[nt] = wmma_bf16(a.v, bfr.v, acc[nt]);
    }
  }

#pragma unroll
  for (int nt = 0; nt < 4; ++nt) {
    const int ncol = n0 + nt * 16 + lm;
    const float bv = bias[ncol];
#pragma unroll
    for (int i = 0; i < 8; ++i) {
      int r = m0 + w * 16 + i + ((l >> 4) << 3);
      out[(size_t)r * D + ncol] = acc[nt][i] + bv;
    }
  }
}

// =====================================================================
// Flash attention: per (b,h, 64-row q tile). 4 waves x 16 q rows.
// Q,K,V in [B,H,S,DK] bf16; output Xb in [B,S,D] bf16.
// =====================================================================
constexpr int KST = 72;   // padded LDS row stride for 64-wide tiles

__global__ __launch_bounds__(128)
void attn(const unsigned short* __restrict__ Qb, const unsigned short* __restrict__ Kb,
          const unsigned short* __restrict__ Vb, const int* __restrict__ maskp,
          unsigned short* __restrict__ Xb) {
  __shared__ __attribute__((aligned(16))) unsigned short Ks[64 * KST];       // [key][dk]
  __shared__ __attribute__((aligned(16))) unsigned short Vt[64 * KST];       // [dk][key]
  __shared__ __attribute__((aligned(16))) unsigned short Ps[4 * 16 * KST];   // per-wave P

  const int tid  = threadIdx.x;
  const int w    = tid >> 5;
  const int l    = tid & 31;
  const int lm   = l & 15;
  const int kh8  = (l >> 4) * 8;
  const int kb16 = (l >> 4) * 16;
  const int qtile = blockIdx.x;          // 0..S/64-1
  const int bh    = blockIdx.y;          // 0..B*H-1
  const int b     = bh >> 4;             // H == 16
  const int h     = bh & 15;
  const int qbase = qtile * 64 + w * 16;
  const float scale = 0.125f;            // 1/sqrt(DK)

  // Q fragments (dk 0..31 and 32..63) straight from global in A-frag layout
  const unsigned short* qrow = Qb + (((size_t)bh) * S + qbase + lm) * DK;
  Frag32B qf0, qf1;
  qf0.q[0] = *(const u32x4*)(qrow + kh8);
  qf0.q[1] = *(const u32x4*)(qrow + 16 + kh8);
  qf1.q[0] = *(const u32x4*)(qrow + 32 + kh8);
  qf1.q[1] = *(const u32x4*)(qrow + 48 + kh8);

  v8f oacc[4];
  const v8f zf = {0.f,0.f,0.f,0.f,0.f,0.f,0.f,0.f};
#pragma unroll
  for (int nt = 0; nt < 4; ++nt) oacc[nt] = zf;
  float mrow[8], lrow[8];
#pragma unroll
  for (int i = 0; i < 8; ++i) { mrow[i] = -1e30f; lrow[i] = 0.f; }

  for (int kt = 0; kt < S / 64; ++kt) {
    __syncthreads();
    const size_t kvbase = (((size_t)bh) * S + (size_t)kt * 64) * DK;
    // K tile row-major; V tile transposed on LDS store
#pragma unroll
    for (int i = 0; i < 4; ++i) {
      int c   = tid + i * 128;            // 0..511 chunks of 8 bf16
      int key = c >> 3;
      int dk8 = (c & 7) * 8;
      u32x4 kv = *(const u32x4*)(Kb + kvbase + (size_t)key * DK + dk8);
      *(u32x4*)&Ks[key * KST + dk8] = kv;
      Chunk16B vv;
      vv.q = *(const u32x4*)(Vb + kvbase + (size_t)key * DK + dk8);
#pragma unroll
      for (int j = 0; j < 8; ++j) Vt[(dk8 + j) * KST + key] = vv.h[j];
    }
    if (kt + 1 < S / 64) {   // CDNA5 global_prefetch_b8 for next tile
      __builtin_prefetch(Kb + kvbase + (size_t)64 * DK + (size_t)tid * 32, 0, 0);
      __builtin_prefetch(Vb + kvbase + (size_t)64 * DK + (size_t)tid * 32, 0, 0);
    }
    __syncthreads();

    // scores S = Q K^T  (4 column tiles of 16 keys, 2 k-steps of dk)
    v8f sc[4];
#pragma unroll
    for (int nt = 0; nt < 4; ++nt) {
      const int krow = nt * 16 + lm;
      Frag32B kf0, kf1;
      kf0.q[0] = *(const u32x4*)&Ks[krow * KST + kh8];
      kf0.q[1] = *(const u32x4*)&Ks[krow * KST + 16 + kh8];
      kf1.q[0] = *(const u32x4*)&Ks[krow * KST + 32 + kh8];
      kf1.q[1] = *(const u32x4*)&Ks[krow * KST + 48 + kh8];
      v8f s = zf;
      s = wmma_bf16(qf0.v, kf0.v, s);
      s = wmma_bf16(qf1.v, kf1.v, s);
      const int kidx = kt * 64 + nt * 16 + lm;
      const int mv = maskp[b * S + kidx];
#pragma unroll
      for (int i = 0; i < 8; ++i) sc[nt][i] = mv ? s[i] * scale : -1e9f;
    }

    // online softmax (row = i + 8*(lane>=16); column = lane&15)
    float rm[8];
#pragma unroll
    for (int i = 0; i < 8; ++i)
      rm[i] = rmax16(fmaxf(fmaxf(sc[0][i], sc[1][i]), fmaxf(sc[2][i], sc[3][i])));
    float mn[8], corr[8], ps[8];
#pragma unroll
    for (int i = 0; i < 8; ++i) {
      mn[i] = fmaxf(mrow[i], rm[i]);
      corr[i] = __expf(mrow[i] - mn[i]);
      mrow[i] = mn[i];
      ps[i] = 0.f;
    }
#pragma unroll
    for (int nt = 0; nt < 4; ++nt) {
#pragma unroll
      for (int i = 0; i < 8; ++i) {
        float p = __expf(sc[nt][i] - mn[i]);
        ps[i] += p;
        int row = i + ((l >> 4) << 3);
        Ps[(w * 16 + row) * KST + nt * 16 + lm] = f2bf(p);
      }
    }
#pragma unroll
    for (int i = 0; i < 8; ++i) lrow[i] = lrow[i] * corr[i] + rsum16(ps[i]);
#pragma unroll
    for (int nt = 0; nt < 4; ++nt)
#pragma unroll
      for (int i = 0; i < 8; ++i) oacc[nt][i] *= corr[i];
    __syncthreads();   // publish P before re-reading in A-frag layout

    // O += P V   (A = P 16x64 via LDS; B = Vt contiguous keys)
    Frag32B pf0, pf1;
    const unsigned short* prow = &Ps[(w * 16 + lm) * KST];
    pf0.q[0] = *(const u32x4*)(prow + kh8);
    pf0.q[1] = *(const u32x4*)(prow + 16 + kh8);
    pf1.q[0] = *(const u32x4*)(prow + 32 + kh8);
    pf1.q[1] = *(const u32x4*)(prow + 48 + kh8);
#pragma unroll
    for (int nt = 0; nt < 4; ++nt) {
      const int vcol = nt * 16 + lm;   // dk column
      Frag32B vf0, vf1;
      vf0.q[0] = *(const u32x4*)&Vt[vcol * KST + kh8];
      vf0.q[1] = *(const u32x4*)&Vt[vcol * KST + 16 + kh8];
      vf1.q[0] = *(const u32x4*)&Vt[vcol * KST + 32 + kh8];
      vf1.q[1] = *(const u32x4*)&Vt[vcol * KST + 48 + kh8];
      oacc[nt] = wmma_bf16(pf0.v, vf0.v, oacc[nt]);
      oacc[nt] = wmma_bf16(pf1.v, vf1.v, oacc[nt]);
    }
  }

  // normalize and write to [B,S,D] bf16
#pragma unroll
  for (int nt = 0; nt < 4; ++nt) {
#pragma unroll
    for (int i = 0; i < 8; ++i) {
      float o = oacc[nt][i] / lrow[i];
      int row = qbase + i + ((l >> 4) << 3);
      int d   = h * DK + nt * 16 + lm;
      Xb[((size_t)b * S + row) * D + d] = f2bf(o);
    }
  }
}

// =====================================================================
// Launcher
// =====================================================================
extern "C" void kernel_launch(void* const* d_in, const int* in_sizes, int n_in,
                              void* d_out, int out_size, void* d_ws, size_t ws_size,
                              hipStream_t stream) {
  (void)in_sizes; (void)n_in; (void)out_size; (void)ws_size;
  const float* query = (const float*)d_in[0];
  const float* key   = (const float*)d_in[1];
  const float* value = (const float*)d_in[2];
  const int*   mask  = (const int*)d_in[3];
  const float* Wq = (const float*)d_in[4];
  const float* bq = (const float*)d_in[5];
  const float* Wk = (const float*)d_in[6];
  const float* bk = (const float*)d_in[7];
  const float* Wv = (const float*)d_in[8];
  const float* bv = (const float*)d_in[9];
  const float* Wo = (const float*)d_in[10];
  const float* bo = (const float*)d_in[11];

  const size_t nElem = (size_t)B * S * D;           // 4,194,304
  unsigned short* Qb = (unsigned short*)d_ws;       // bf16 Q [B,H,S,DK]
  unsigned short* Kb = Qb + nElem;                  // bf16 K
  unsigned short* Vb = Kb + nElem;                  // bf16 V
  unsigned short* Xb = Vb + nElem;                  // bf16 attention output [B,S,D]

  dim3 gg(D / 64, (B * S) / 64);                    // (16, 64)
  dim3 gb(128);
  gemm_qkv<<<gg, gb, 0, stream>>>(query, Wq, bq, Qb);
  gemm_qkv<<<gg, gb, 0, stream>>>(key,   Wk, bk, Kb);
  gemm_qkv<<<gg, gb, 0, stream>>>(value, Wv, bv, Vb);

  attn<<<dim3(S / 64, B * H), gb, 0, stream>>>(Qb, Kb, Vb, mask, Xb);

  gemm_out<<<gg, gb, 0, stream>>>(Xb, Wo, bo, (float*)d_out);
}
